// CrossAttnNoQueryProj_16312285790982
// MI455X (gfx1250) — matrix operand order repaired and checked by reading, not compiled, not run on hardware
//
#include <hip/hip_runtime.h>

// ---------------------------------------------------------------------------
// Cross-attention (no query projection) for MI455X / gfx1250, bf16 WMMA path.
// B=8, Nq=Nkv=1024, D=2304, H=8, hd=288.
// Projections + flash attention, with async global->LDS staging (ASYNCcnt)
// and double-buffered GEMM tiles.
// ---------------------------------------------------------------------------

typedef __bf16 bf16;
typedef __bf16 v16bf __attribute__((ext_vector_type(16)));
typedef __bf16 v8bf  __attribute__((ext_vector_type(8)));
typedef float  v8f   __attribute__((ext_vector_type(8)));

#define BATCH 8
#define NQ    1024
#define NKV   1024
#define DM    2304
#define NH    8
#define HD    288            // = 18 * 16

__device__ __forceinline__ v16bf combine16(v8bf lo, v8bf hi) {
  return __builtin_shufflevector(lo, hi, 0,1,2,3,4,5,6,7,8,9,10,11,12,13,14,15);
}

// async 16B copy: global -> LDS, tracked by ASYNCcnt
__device__ __forceinline__ void async_cp16(const bf16* g, const bf16* l) {
  uint32_t loff = (uint32_t)(uintptr_t)l;          // wave-relative LDS byte addr
  uint64_t gaddr = (uint64_t)(uintptr_t)g;
  asm volatile("global_load_async_to_lds_b128 %0, %1, off"
               :: "v"(loff), "v"(gaddr) : "memory");
}
__device__ __forceinline__ void wait_async0() {
  asm volatile("s_wait_asynccnt 0" ::: "memory");
}

// ---------------------------------------------------------------------------
// fp32 -> bf16 conversion (optionally scaled), 4 elements per thread
// ---------------------------------------------------------------------------
__global__ void cvt_bf16_kernel(const float* __restrict__ in,
                                bf16* __restrict__ out,
                                long n4, float scale) {
  long i = (long)blockIdx.x * blockDim.x + threadIdx.x;
  if (i < n4) {
    float4 f = ((const float4*)in)[i];
    long o = i * 4;
    out[o + 0] = (bf16)(f.x * scale);
    out[o + 1] = (bf16)(f.y * scale);
    out[o + 2] = (bf16)(f.z * scale);
    out[o + 3] = (bf16)(f.w * scale);
  }
}

// ---------------------------------------------------------------------------
// NT GEMM: C[M,N] = A[M,K] * W[N,K]^T   (bf16 in/out, f32 accum)
// Block tile 128x128, 8 waves (4x2), wave tile 32x64 = 2x4 WMMA tiles.
// A/B 128x32 K-slabs staged in LDS via async-to-LDS, double buffered:
// 16 KB of L2 traffic per block K-step for 64 WMMAs (~64 FLOP/byte from L2).
// LDS rows padded to 40 elems (20 banks) -> conflict-free 16-lane frag reads.
// ---------------------------------------------------------------------------
#define LP 40   // padded LDS row stride (elements) for a 32-element row

__global__ __launch_bounds__(256)
void gemm_nt_bf16(const bf16* __restrict__ A, const bf16* __restrict__ W,
                  bf16* __restrict__ C, int M, int N, int K) {
  __shared__ bf16 As[2][128 * LP];
  __shared__ bf16 Bs[2][128 * LP];

  const int tid  = threadIdx.x;
  const int lane = tid & 31;
  const int wave = tid >> 5;
  const int wm = wave & 3;          // 4 waves along M
  const int wn = wave >> 2;         // 2 waves along N
  const int lm = lane & 15;
  const int lh = lane >> 4;
  const size_t mb = (size_t)blockIdx.x * 128;
  const size_t nb = (size_t)blockIdx.y * 128;

  // stage the two 128x32 slabs for K-offset k0 into buffer `buf`
  auto stage = [&](int buf, int k0) {
    // 128 rows x 4 16B-segments = 512 segs per slab; 2 per thread per slab
#pragma unroll
    for (int s = 0; s < 2; ++s) {
      int seg = tid + s * 256;
      int r  = seg >> 2;
      int c8 = (seg & 3) * 8;
      async_cp16(A + (mb + r) * K + k0 + c8, &As[buf][r * LP + c8]);
      async_cp16(W + (nb + r) * K + k0 + c8, &Bs[buf][r * LP + c8]);
    }
  };

  v8f acc[2][4] = {};
  const int NSTEP = K / 32;
  stage(0, 0);

  for (int step = 0; step < NSTEP; ++step) {
    const int buf = step & 1;
    wait_async0();          // slab `buf` landed (this wave's copies)
    __syncthreads();        // ...and everyone else's; prev readers done
    if (step + 1 < NSTEP) stage(buf ^ 1, (step + 1) * 32);

    v16bf a[2], b[4];
#pragma unroll
    for (int i = 0; i < 2; ++i) {
      // A 16x32 frag: lanes 0-15 K=0..7,16..23 ; lanes 16-31 K=8..15,24..31
      const bf16* p = &As[buf][(wm * 32 + i * 16 + lm) * LP + lh * 8];
      a[i] = combine16(*(const v8bf*)p, *(const v8bf*)(p + 16));
    }
#pragma unroll
    for (int j = 0; j < 4; ++j) {
      // B 32x16 frag: lane n = column n; lanes 0-15 K=0..7,16..23 ; lanes 16-31 K=8..15,24..31
      const bf16* p = &Bs[buf][(wn * 64 + j * 16 + lm) * LP + lh * 8];
      b[j] = combine16(*(const v8bf*)p, *(const v8bf*)(p + 16));
    }
#pragma unroll
    for (int i = 0; i < 2; ++i)
#pragma unroll
      for (int j = 0; j < 4; ++j)
        acc[i][j] = __builtin_amdgcn_wmma_f32_16x16x32_bf16(
            false, a[i], false, b[j], (short)0, acc[i][j], false, false);
    __syncthreads();        // all waves done reading slab `buf`
  }

  // C/D layout: lane n = lm ; VGPR r -> row r + 8*lh
#pragma unroll
  for (int i = 0; i < 2; ++i)
#pragma unroll
    for (int j = 0; j < 4; ++j)
#pragma unroll
      for (int r = 0; r < 8; ++r) {
        size_t m = mb + wm * 32 + i * 16 + r + 8 * lh;
        size_t n = nb + wn * 64 + j * 16 + lm;
        C[m * N + n] = (bf16)acc[i][j][r];
      }
}

// ---------------------------------------------------------------------------
// Fused flash attention: block = (b, h, 128 query rows), 8 waves x 16 rows.
// Q tile held in registers (9 x v16bf). Per 32-key chunk:
//  - K chunk (32x288) async-staged to LDS, shared by all 8 waves
//  - V chunk transposed into LDS so PV B-fragments are contiguous
//  - S via 9 WMMA K-steps; mask + online softmax (shfl_xor row reductions)
//  - P staged per-wave through LDS (C-layout -> A-layout), s_wait_dscnt
//  - O: 18 f32 WMMA accumulators (16x288)
// ---------------------------------------------------------------------------
#define KCP 296   // Kc padded row stride (288 + 8): 148 banks/row -> no conflicts
#define VTP 40    // Vt padded row stride (32 + 8)

__global__ __launch_bounds__(256)
void attn_kernel(const bf16* __restrict__ q,   // (B*NQ, DM) pre-scaled bf16
                 const bf16* __restrict__ kp,  // (B*NKV, DM) bf16
                 const bf16* __restrict__ vp,  // (B*NKV, DM) bf16
                 const unsigned char* __restrict__ mask, // (B, NKV) bool
                 float* __restrict__ out) {    // (B*NQ, DM) f32
  __shared__ bf16 Kc[32 * KCP];       // K chunk, natural layout  (~18.5 KB)
  __shared__ bf16 Vt[HD * VTP];       // V chunk transposed [e][kv] (~22.5 KB)
  __shared__ bf16 Plds[8][16 * 32];   // per-wave P staging (8 KB)

  const int tid  = threadIdx.x;
  const int lane = tid & 31;
  const int wave = tid >> 5;
  const int lm = lane & 15;
  const int lh = lane >> 4;
  const int b  = blockIdx.z;
  const int h  = blockIdx.y;
  const int q0 = blockIdx.x * 128;

  const size_t qrow = (size_t)b * NQ + q0 + wave * 16;
  const bf16* qbase = q  + qrow * DM + h * HD;
  const bf16* kbase = kp + (size_t)b * NKV * DM + h * HD;
  const bf16* vbase = vp + (size_t)b * NKV * DM + h * HD;
  const unsigned char* mb = mask + b * NKV;

  // Q tile resident in registers: 9 A-fragments (16 x 288 total)
  v16bf qf[9];
#pragma unroll
  for (int dd = 0; dd < 9; ++dd) {
    const bf16* qp = qbase + (size_t)lm * DM + dd * 32 + lh * 8;
    qf[dd] = combine16(*(const v8bf*)qp, *(const v8bf*)(qp + 16));
  }

  v8f o[18] = {};
  float mrow[8], lrow[8];
#pragma unroll
  for (int j = 0; j < 8; ++j) { mrow[j] = -3.0e38f; lrow[j] = 0.0f; }

  for (int kv0 = 0; kv0 < NKV; kv0 += 32) {
    // ---- async-stage K chunk (32 rows x 576B = 1152 16B segments) ----
    for (int seg = tid; seg < 32 * (HD / 8); seg += 256) {
      int kv = seg / (HD / 8);
      int c8 = (seg % (HD / 8)) * 8;
      async_cp16(kbase + (size_t)(kv0 + kv) * DM + c8, &Kc[kv * KCP + c8]);
    }
    // ---- cooperative transpose of V chunk into LDS: Vt[e*VTP + kv] ----
    for (int seg = tid; seg < (HD * 32) / 8; seg += 256) {
      int kv = seg / (HD / 8);
      int e8 = (seg % (HD / 8)) * 8;
      v8bf d = *(const v8bf*)(vbase + (size_t)(kv0 + kv) * DM + e8);
#pragma unroll
      for (int u = 0; u < 8; ++u) Vt[(e8 + u) * VTP + kv] = d[u];
    }
    const bool msk0 = mb[kv0 + lm]      != 0;
    const bool msk1 = mb[kv0 + 16 + lm] != 0;
    wait_async0();
    __syncthreads();

    // ---- S = Q_tile (16x288) @ K_chunk^T (288x32): 9 WMMA K-steps ----
    v8f s0 = {}, s1 = {};
#pragma unroll
    for (int dd = 0; dd < 9; ++dd) {
      const bf16* p0 = &Kc[lm        * KCP + dd * 32 + lh * 8];
      const bf16* p1 = &Kc[(16 + lm) * KCP + dd * 32 + lh * 8];
      v16bf b0 = combine16(*(const v8bf*)p0, *(const v8bf*)(p0 + 16));
      v16bf b1 = combine16(*(const v8bf*)p1, *(const v8bf*)(p1 + 16));
      s0 = __builtin_amdgcn_wmma_f32_16x16x32_bf16(false, qf[dd], false, b0, (short)0, s0, false, false);
      s1 = __builtin_amdgcn_wmma_f32_16x16x32_bf16(false, qf[dd], false, b1, (short)0, s1, false, false);
    }

    // ---- mask + online softmax (rows live across 16-lane halves) ----
    float scalev[8];
#pragma unroll
    for (int j = 0; j < 8; ++j) {
      float a0 = msk0 ? -3.0e38f : s0[j];
      float a1 = msk1 ? -3.0e38f : s1[j];
      float rm = fmaxf(a0, a1);
      rm = fmaxf(rm, __shfl_xor(rm, 1, 32));
      rm = fmaxf(rm, __shfl_xor(rm, 2, 32));
      rm = fmaxf(rm, __shfl_xor(rm, 4, 32));
      rm = fmaxf(rm, __shfl_xor(rm, 8, 32));
      float mnew = fmaxf(mrow[j], rm);
      float sc   = __expf(mrow[j] - mnew);
      float p0   = __expf(a0 - mnew);
      float p1   = __expf(a1 - mnew);
      float rs = p0 + p1;
      rs += __shfl_xor(rs, 1, 32);
      rs += __shfl_xor(rs, 2, 32);
      rs += __shfl_xor(rs, 4, 32);
      rs += __shfl_xor(rs, 8, 32);
      lrow[j] = lrow[j] * sc + rs;
      mrow[j] = mnew;
      scalev[j] = sc;
      // C-layout (row j+8*lh, col lm) -> A-layout staging in LDS
      Plds[wave][(j + 8 * lh) * 32 + lm]      = (bf16)p0;
      Plds[wave][(j + 8 * lh) * 32 + 16 + lm] = (bf16)p1;
    }

    // wave-private LDS round trip: drain the DS counter only
    asm volatile("s_wait_dscnt 0" ::: "memory");

    // reload P as a 16x32 A-fragment
    const bf16* pb = &Plds[wave][0];
    v8bf plo = *(const v8bf*)(pb + lm * 32 + lh * 8);
    v8bf phi = *(const v8bf*)(pb + lm * 32 + 16 + lh * 8);
    v16bf pa = combine16(plo, phi);

    // ---- O = O * scale + P (16x32) @ V_chunk (32x288): 18 WMMAs ----
#pragma unroll
    for (int t = 0; t < 18; ++t) {
#pragma unroll
      for (int j = 0; j < 8; ++j) o[t][j] *= scalev[j];
      const bf16* vtp = &Vt[(t * 16 + lm) * VTP + lh * 8];
      v16bf bv = combine16(*(const v8bf*)vtp, *(const v8bf*)(vtp + 16));
      o[t] = __builtin_amdgcn_wmma_f32_16x16x32_bf16(
          false, pa, false, bv, (short)0, o[t], false, false);
    }
    __syncthreads();  // protect Kc/Vt before the next chunk overwrites them
  }

  // ---- epilogue: divide by softmax denominator, write f32 output ----
#pragma unroll
  for (int t = 0; t < 18; ++t)
#pragma unroll
    for (int j = 0; j < 8; ++j) {
      int m = j + 8 * lh;
      out[(qrow + m) * DM + h * HD + t * 16 + lm] = o[t][j] / lrow[j];
    }
}

// ---------------------------------------------------------------------------
// Host launcher
// inputs: 0=q f32, 1=k f32, 2=v f32, 3=key_padding_mask bool, 4=Wk f32, 5=Wv f32
// output: x (B, Nq, D) f32
// ---------------------------------------------------------------------------
extern "C" void kernel_launch(void* const* d_in, const int* in_sizes, int n_in,
                              void* d_out, int out_size, void* d_ws, size_t ws_size,
                              hipStream_t stream) {
  const float* q  = (const float*)d_in[0];
  const float* k  = (const float*)d_in[1];
  const float* v  = (const float*)d_in[2];
  const unsigned char* mask = (const unsigned char*)d_in[3];
  const float* Wk = (const float*)d_in[4];
  const float* Wv = (const float*)d_in[5];
  float* out = (float*)d_out;

  const size_t nx = (size_t)BATCH * NQ * DM;   // 18,874,368 elements
  const size_t nw = (size_t)DM * DM;           //  5,308,416 elements

  // workspace layout (bf16): kp | vp | xb (staging: k, then v, then q) | wb
  bf16* kp = (bf16*)d_ws;
  bf16* vp = kp + nx;
  bf16* xb = vp + nx;
  bf16* wb = xb + nx;

  const int cvt_blk = 256;
  const int gx_x = (int)((nx / 4 + cvt_blk - 1) / cvt_blk);
  const int gx_w = (int)((nw / 4 + cvt_blk - 1) / cvt_blk);
  const dim3 ggemm(8 * NQ / 128, DM / 128);    // (64, 18)
  const dim3 gattn(NQ / 128, NH, BATCH);       // (8, 8, 8)

  // K projection
  cvt_bf16_kernel<<<gx_x, cvt_blk, 0, stream>>>(k,  xb, (long)(nx / 4), 1.0f);
  cvt_bf16_kernel<<<gx_w, cvt_blk, 0, stream>>>(Wk, wb, (long)(nw / 4), 1.0f);
  gemm_nt_bf16<<<ggemm, 256, 0, stream>>>(xb, wb, kp, BATCH * NKV, DM, DM);

  // V projection
  cvt_bf16_kernel<<<gx_x, cvt_blk, 0, stream>>>(v,  xb, (long)(nx / 4), 1.0f);
  cvt_bf16_kernel<<<gx_w, cvt_blk, 0, stream>>>(Wv, wb, (long)(nw / 4), 1.0f);
  gemm_nt_bf16<<<ggemm, 256, 0, stream>>>(xb, wb, vp, BATCH * NKV, DM, DM);

  // Q: pre-scale by hd^-0.5 during conversion
  const float qscale = 1.0f / sqrtf((float)HD);
  cvt_bf16_kernel<<<gx_x, cvt_blk, 0, stream>>>(q, xb, (long)(nx / 4), qscale);

  // fused attention
  attn_kernel<<<gattn, 256, 0, stream>>>(xb, kp, vp, mask, out);
}
